// MLPBranch_59596966199730
// MI455X (gfx1250) — matrix operand order repaired and checked by reading, not compiled, not run on hardware
//
#include <hip/hip_runtime.h>

// ---------------------------------------------------------------------------
// MLPBranch: out[b,s,e] = patches[b,s,:] . (weight[branch[b]] * mask)[e,:] + bias[branch[b],e]
// B=128, S=2048, E=512, W=128, branches=5, windows={8,16,32,64,128}
//
// Strategy: fold mask into masked-f16 weight copy (prepass, L2 resident),
// then one f16->f32 WMMA GEMM per sample tile. Memory-bound target ~29us.
// ---------------------------------------------------------------------------

typedef _Float16 v16h __attribute__((ext_vector_type(16)));
typedef float    v8f  __attribute__((ext_vector_type(8)));
typedef float    v4f  __attribute__((ext_vector_type(4)));

#define DIM_S 2048
#define DIM_E 512
#define DIM_W 128
#define N_BRANCH 5

// ---------------------------------------------------------------------------
// Prepass: masked f16 weights.  wf[branch][e][w] = w < win[branch] ? f16(weight) : 0
// ---------------------------------------------------------------------------
__global__ __launch_bounds__(256)
void prep_weights_kernel(const float* __restrict__ weight,
                         _Float16* __restrict__ wf, int total) {
    int idx = blockIdx.x * blockDim.x + threadIdx.x;
    if (idx >= total) return;
    const int wins[N_BRANCH] = {8, 16, 32, 64, 128};
    int w      = idx & (DIM_W - 1);
    int branch = idx >> 16;                  // E*W = 512*128 = 65536 per branch
    float v = weight[idx];
    wf[idx] = (w < wins[branch]) ? (_Float16)v : (_Float16)0.0f;
}

// ---------------------------------------------------------------------------
// Main GEMM.
// Grid: B * (S/128) * (E/128) = 128 * 16 * 4 = 8192 blocks, 256 threads (8 waves).
// Workgroup tile: 128(S) x 128(E).  Wave tile: 32(S) x 64(E) = 2x4 WMMA frags.
// K loop: 4 steps of 32 (f16 WMMA K).
// ---------------------------------------------------------------------------
__global__ __launch_bounds__(256)
void mlp_branch_gemm_kernel(const float* __restrict__ patches,
                            const int* __restrict__ branch_idx,
                            const _Float16* __restrict__ wf,
                            const float* __restrict__ bias,
                            float* __restrict__ out) {
    const int blk  = blockIdx.x;
    const int nblk = blk & 3;              // E/128 = 4
    const int mblk = (blk >> 2) & 15;      // S/128 = 16
    const int b    = blk >> 6;             // sample

    const int lane = threadIdx.x & 31;
    const int wave = threadIdx.x >> 5;
    const int waveM = (wave & 3) * 32;     // 4 waves along S
    const int waveN = (wave >> 2) * 64;    // 2 waves along E

    const int branch = branch_idx[b];

    const float*    Abase = patches + ((size_t)b * DIM_S + (size_t)(mblk * 128 + waveM)) * DIM_W;
    const _Float16* Bbase = wf + (size_t)branch * DIM_E * DIM_W
                               + (size_t)(nblk * 128 + waveN) * DIM_W;
    const float*    biasp = bias + branch * DIM_E + nblk * 128 + waveN;

    // --- A fragment addressing (16-bit A 16x32, contiguous k order, same as B):
    //   lanes 0-15 : row M=lane,    K = kb..kb+15 with kb=0
    //   lanes16-31 : row M=lane-16, same with kb=16
    const int r  = lane & 15;
    const int kb = (lane >> 4) << 4;       // 0 or 16
    // --- B fragment addressing (16-bit B 32x16, row striped across lanes):
    //   lane n holds column n; lanes 0-15: K=0..15, lanes 16-31: K=16..31
    const int kh = (lane >> 4) << 4;       // 0 or 16

    v8f acc[2][4];
#pragma unroll
    for (int mi = 0; mi < 2; ++mi)
#pragma unroll
        for (int ni = 0; ni < 4; ++ni)
            acc[mi][ni] = (v8f){0.f, 0.f, 0.f, 0.f, 0.f, 0.f, 0.f, 0.f};

#pragma unroll
    for (int ks = 0; ks < 4; ++ks) {
        const int ko = ks * 32;

        // Load + convert A fragments (fp32 memory -> f16 regs)
        v16h afrag[2];
#pragma unroll
        for (int mi = 0; mi < 2; ++mi) {
            const float* p = Abase + (size_t)(mi * 16 + r) * DIM_W + kb + ko;
            v4f x0 = *(const v4f*)(p);
            v4f x1 = *(const v4f*)(p + 4);
            v4f x2 = *(const v4f*)(p + 8);
            v4f x3 = *(const v4f*)(p + 12);
            v16h f;
#pragma unroll
            for (int i = 0; i < 4; ++i) {
                f[i]      = (_Float16)x0[i];
                f[4 + i]  = (_Float16)x1[i];
                f[8 + i]  = (_Float16)x2[i];
                f[12 + i] = (_Float16)x3[i];
            }
            afrag[mi] = f;
        }

        // Load B fragments: weight[e][w] row-major == B^T, so lane n reads
        // 16 contiguous f16 of its column -> single 32B load, 32B aligned.
        v16h bfrag[4];
#pragma unroll
        for (int ni = 0; ni < 4; ++ni) {
            const _Float16* q = Bbase + (size_t)(ni * 16 + r) * DIM_W + kh + ko;
            bfrag[ni] = *(const v16h*)q;
        }

        // 8 independent WMMAs per K step
#pragma unroll
        for (int mi = 0; mi < 2; ++mi)
#pragma unroll
            for (int ni = 0; ni < 4; ++ni)
                acc[mi][ni] = __builtin_amdgcn_wmma_f32_16x16x32_f16(
                    /*neg_a=*/false, afrag[mi],
                    /*neg_b=*/false, bfrag[ni],
                    /*c_mod=*/(short)0, acc[mi][ni],
                    /*reuse_a=*/false, /*reuse_b=*/false);
    }

    // Epilogue: C/D layout -> VGPR rr holds M=rr (lanes 0-15) / M=rr+8 (16-31),
    // N = lane&15. Add bias (depends only on E) and store fp32.
    const int mo    = (lane >> 4) * 8;
    const int n_off = lane & 15;
    const size_t outbase = ((size_t)b * DIM_S + (size_t)(mblk * 128 + waveM)) * DIM_E
                         + (size_t)(nblk * 128 + waveN);

#pragma unroll
    for (int ni = 0; ni < 4; ++ni) {
        const float bv = biasp[ni * 16 + n_off];
#pragma unroll
        for (int mi = 0; mi < 2; ++mi) {
#pragma unroll
            for (int rr = 0; rr < 8; ++rr) {
                const int m = mi * 16 + mo + rr;
                out[outbase + (size_t)m * DIM_E + ni * 16 + n_off] = acc[mi][ni][rr] + bv;
            }
        }
    }
}

// ---------------------------------------------------------------------------
extern "C" void kernel_launch(void* const* d_in, const int* in_sizes, int n_in,
                              void* d_out, int out_size, void* d_ws, size_t ws_size,
                              hipStream_t stream) {
    const float* patches    = (const float*)d_in[0];   // (128, 2048, 128) f32
    const int*   branch_idx = (const int*)d_in[1];     // (128,) i32
    const float* weight     = (const float*)d_in[2];   // (5, 512, 128) f32
    const float* bias       = (const float*)d_in[3];   // (5, 512) f32
    float*       out        = (float*)d_out;           // (128, 2048, 512) f32

    _Float16* wf = (_Float16*)d_ws;                    // masked f16 weights, 655 KB

    const int total_w = N_BRANCH * DIM_E * DIM_W;      // 327680
    prep_weights_kernel<<<(total_w + 255) / 256, 256, 0, stream>>>(weight, wf, total_w);

    const int grid = 128 * (DIM_S / 128) * (DIM_E / 128);  // 8192
    mlp_branch_gemm_kernel<<<grid, 256, 0, stream>>>(patches, branch_idx, wf, bias, out);
}
